// GlobalNodeInductiveActorNetwork_83734682403381
// MI455X (gfx1250) — compile-verified, matrix operand order checked by reading, not run on hardware
//
#include <hip/hip_runtime.h>
#include <hip/hip_bf16.h>
#include <math.h>

typedef __attribute__((ext_vector_type(16))) __bf16 v16bf;
typedef __attribute__((ext_vector_type(8)))  __bf16 v8bf;
typedef __attribute__((ext_vector_type(4)))  __bf16 v4bf;
typedef __attribute__((ext_vector_type(8)))  float  v8f;

#define NN 57344
#define EE 458752
#define INF_NEG (-INFINITY)

// ---------------------------------------------------------------------------
// WMMA bf16 GEMM:  C[M,N] = A[M,K] @ W[K,N]  (+bias) (+relu)
// Block tile 128x64, 8 waves, each wave a 32x32 patch = 2x2 v_wmma tiles.
// Double-buffered LDS; As row-major [128][40], Bt TRANSPOSED [64][40] so every
// fragment is two contiguous aligned 16-byte ds_load_b128 per lane.
// Pipeline per k-step: issue next-tile global loads -> ds frag loads ->
// 4x v_wmma -> cvt+store next tile -> one barrier.
// M % 128 == 0, N % 64 == 0, K % 32 == 0 at all call sites.
// flags: bit0 = relu, bit1 = add bias
// ---------------------------------------------------------------------------
#define LDS_STRIDE 40   // 32 + 8 pad (bf16); 80 B row pitch, 16B aligned

__global__ __launch_bounds__(256) void wmma_gemm(
    const float* __restrict__ A, const float* __restrict__ W,
    const float* __restrict__ bias, float* __restrict__ C,
    int M, int N, int K, int flags)
{
    __shared__ __align__(16) __bf16 As[2][128 * LDS_STRIDE];
    __shared__ __align__(16) __bf16 Bt[2][64 * LDS_STRIDE];

    const int tid  = threadIdx.x;
    const int lane = tid & 31;
    const int wave = tid >> 5;
    const int hl   = lane & 15;
    const int hi   = lane >> 4;

    const int wm = (wave & 3) * 32;   // wave row offset inside 128
    const int wn = (wave >> 2) * 32;  // wave col offset inside 64

    const long rowBase = (long)blockIdx.x * 128;
    const int  colBase = blockIdx.y * 64;

    // ---- tile fetch (registers only; issues 4x b128 + 8x b32, no waits) ----
    auto load_tile = [&](int kk, float4* fa, float* fv) {
        #pragma unroll
        for (int i = 0; i < 4; ++i) {
            int q = tid + i * 256;                 // 1024 quads of A
            int r = q >> 3, c = (q & 7) * 4;
            fa[i] = *(const float4*)(A + (rowBase + r) * K + kk + c);
        }
        #pragma unroll
        for (int hf = 0; hf < 2; ++hf) {
            int n  = tid & 63;
            int kq = ((tid >> 6) << 2) + hf * 16;  // 0,4,8,12 / 16,20,24,28
            const float* wp = W + (long)(kk + kq) * N + colBase + n;
            fv[hf * 4 + 0] = wp[0];
            fv[hf * 4 + 1] = wp[N];
            fv[hf * 4 + 2] = wp[2 * N];
            fv[hf * 4 + 3] = wp[3 * N];
        }
    };
    // ---- tile commit: fp32 -> bf16 (v_cvt_pk_bf16_f32) -> ds_store_b64 -----
    auto store_tile = [&](int buf, const float4* fa, const float* fv) {
        #pragma unroll
        for (int i = 0; i < 4; ++i) {
            int q = tid + i * 256;
            int r = q >> 3, c = (q & 7) * 4;
            v4bf o;
            o[0] = (__bf16)fa[i].x; o[1] = (__bf16)fa[i].y;
            o[2] = (__bf16)fa[i].z; o[3] = (__bf16)fa[i].w;
            *(v4bf*)&As[buf][r * LDS_STRIDE + c] = o;
        }
        #pragma unroll
        for (int hf = 0; hf < 2; ++hf) {
            int n  = tid & 63;
            int kq = ((tid >> 6) << 2) + hf * 16;
            v4bf o;
            o[0] = (__bf16)fv[hf * 4 + 0]; o[1] = (__bf16)fv[hf * 4 + 1];
            o[2] = (__bf16)fv[hf * 4 + 2]; o[3] = (__bf16)fv[hf * 4 + 3];
            *(v4bf*)&Bt[buf][n * LDS_STRIDE + kq] = o;
        }
    };

    v8f acc[2][2] = {};
    float4 fa[4];
    float  fv[8];

    load_tile(0, fa, fv);
    store_tile(0, fa, fv);
    __syncthreads();

    int cur = 0;
    for (int k0 = 0; k0 < K; k0 += 32) {
        const bool hasNext = (k0 + 32) < K;
        if (hasNext) load_tile(k0 + 32, fa, fv);   // global loads in flight

        // fragments: two aligned 16B LDS loads each (ds_load_b128)
        v16bf af[2], bf_[2];
        #pragma unroll
        for (int ti = 0; ti < 2; ++ti) {
            const __bf16* base = &As[cur][(wm + ti * 16 + hl) * LDS_STRIDE + hi * 8];
            v8bf lo = *(const v8bf*)(base);
            v8bf hh = *(const v8bf*)(base + 16);
            #pragma unroll
            for (int e = 0; e < 8; ++e) { af[ti][e] = lo[e]; af[ti][8 + e] = hh[e]; }
        }
        #pragma unroll
        for (int tj = 0; tj < 2; ++tj) {
            const __bf16* base = &Bt[cur][(wn + tj * 16 + hl) * LDS_STRIDE + hi * 8];
            v8bf lo = *(const v8bf*)(base);
            v8bf hh = *(const v8bf*)(base + 16);
            #pragma unroll
            for (int e = 0; e < 8; ++e) { bf_[tj][e] = lo[e]; bf_[tj][8 + e] = hh[e]; }
        }

        acc[0][0] = __builtin_amdgcn_wmma_f32_16x16x32_bf16(false, af[0], false, bf_[0], (short)0, acc[0][0], false, false);
        acc[0][1] = __builtin_amdgcn_wmma_f32_16x16x32_bf16(false, af[0], false, bf_[1], (short)0, acc[0][1], false, false);
        acc[1][0] = __builtin_amdgcn_wmma_f32_16x16x32_bf16(false, af[1], false, bf_[0], (short)0, acc[1][0], false, false);
        acc[1][1] = __builtin_amdgcn_wmma_f32_16x16x32_bf16(false, af[1], false, bf_[1], (short)0, acc[1][1], false, false);

        if (hasNext) {
            store_tile(cur ^ 1, fa, fv);           // waits loadcnt here only
            __syncthreads();
        }
        cur ^= 1;
    }

    // C/D layout: VGPR r -> row = r + 8*hi, col = hl
    #pragma unroll
    for (int ti = 0; ti < 2; ++ti) {
        #pragma unroll
        for (int tj = 0; tj < 2; ++tj) {
            const int col = colBase + wn + tj * 16 + hl;
            const float bv = (flags & 2) ? bias[col] : 0.0f;
            #pragma unroll
            for (int r = 0; r < 8; ++r) {
                long row = rowBase + wm + ti * 16 + r + hi * 8;
                float v = acc[ti][tj][r] + bv;
                if (flags & 1) v = fmaxf(v, 0.0f);
                C[row * N + col] = v;
            }
        }
    }
}

// ---------------------------------------------------------------------------
// Small utility kernels
// ---------------------------------------------------------------------------
__global__ void fill_kernel(float* p, float v, long count) {
    long i = (long)blockIdx.x * blockDim.x + threadIdx.x;
    if (i < count) p[i] = v;
}

__global__ void compute_bid(const int* __restrict__ batches, int* __restrict__ bid, int n) {
    int i = blockIdx.x * blockDim.x + threadIdx.x;
    if (i >= n) return;
    int b = 0;
    #pragma unroll
    for (int k = 1; k <= 16; ++k) b += (i >= batches[k]) ? 1 : 0;
    bid[i] = b;
}

__global__ void deg_count(const int* __restrict__ ei, float* deg, int E) {
    int e = blockIdx.x * blockDim.x + threadIdx.x;
    if (e < E) atomicAdd(&deg[ei[E + e]], 1.0f);
}

__global__ void deg_finalize(float* dis, int n) {
    int i = blockIdx.x * blockDim.x + threadIdx.x;
    if (i < n) dis[i] = rsqrtf(dis[i]);   // deg >= 1 (self loop) always
}

// ---------------------------------------------------------------------------
// GCN: self-loop init, edge scatter (the memory-bound core), finalize
// ---------------------------------------------------------------------------
__global__ void gcn_self(const float* __restrict__ hw, const float* __restrict__ dis,
                         float* __restrict__ out, int H, long total) {
    long g = (long)blockIdx.x * blockDim.x + threadIdx.x;
    if (g >= total) return;
    int i = (int)(g / H);
    float d = dis[i];
    out[g] = hw[g] * d * d;
}

template <int H>
__global__ __launch_bounds__(256) void gcn_edge(
    const int* __restrict__ ei, const float* __restrict__ hw,
    const float* __restrict__ dis, float* __restrict__ out, int E)
{
    constexpr int EPB = 256 / H;
    int e = blockIdx.x * EPB + threadIdx.x / H;
    int c = threadIdx.x & (H - 1);
    if (e >= E) return;
    int s = ei[e], d = ei[E + e];
    float norm = dis[s] * dis[d];
    atomicAdd(&out[(long)d * H + c], hw[(long)s * H + c] * norm);
}

__global__ void gcn_fin(float* __restrict__ h, const float* __restrict__ b, int H, long total) {
    long g = (long)blockIdx.x * blockDim.x + threadIdx.x;
    if (g >= total) return;
    h[g] = fmaxf(h[g] + b[(int)(g % H)], 0.0f);
}

// ---------------------------------------------------------------------------
// Attention reduce: per-node softmax(att_row) * feat_row, accumulated into
// o[batch, H].  Block = 8 waves * 4 nodes = 32 nodes (never straddles a batch:
// all batch boundaries are multiples of 2048).  LDS accumulation -> one global
// atomic per column per block.
// ---------------------------------------------------------------------------
template <int H>
__global__ __launch_bounds__(256) void attn_reduce(
    const float* __restrict__ att, const float* __restrict__ feat,
    const int* __restrict__ bid, float* __restrict__ o, int n)
{
    __shared__ float acc[H];
    constexpr int R = H / 32;
    const int tid = threadIdx.x, lane = tid & 31, wave = tid >> 5;
    if (tid < H) acc[tid] = 0.0f;
    __syncthreads();

    const int node0 = blockIdx.x * 32;
    const int b = bid[node0];

    for (int s = 0; s < 4; ++s) {
        int node = node0 + wave * 4 + s;
        if (node < n) {
            float v[R];
            float mx = -3.4e38f;
            #pragma unroll
            for (int j = 0; j < R; ++j) {
                v[j] = att[(long)node * H + lane + 32 * j];
                mx = fmaxf(mx, v[j]);
            }
            #pragma unroll
            for (int off = 16; off > 0; off >>= 1) mx = fmaxf(mx, __shfl_xor(mx, off, 32));
            float sum = 0.0f;
            #pragma unroll
            for (int j = 0; j < R; ++j) { v[j] = __expf(v[j] - mx); sum += v[j]; }
            #pragma unroll
            for (int off = 16; off > 0; off >>= 1) sum += __shfl_xor(sum, off, 32);
            float inv = 1.0f / sum;
            #pragma unroll
            for (int j = 0; j < R; ++j) {
                float c = v[j] * inv * feat[(long)node * H + lane + 32 * j];
                atomicAdd(&acc[lane + 32 * j], c);
            }
        }
    }
    __syncthreads();
    if (tid < H) atomicAdd(&o[b * H + tid], acc[tid]);
}

// g_new[b] = g[b] + concat(o[b], g[b]) @ gw + gb     (16 x 256 output, tiny)
__global__ void g_update(const float* __restrict__ o, const float* __restrict__ g,
                         const float* __restrict__ gw, const float* __restrict__ gb,
                         float* __restrict__ gout, int H)
{
    int b = blockIdx.x, j = threadIdx.x;
    int Ktot = H + 256;
    float s = gb[j];
    for (int k = 0; k < Ktot; ++k) {
        float x = (k < H) ? o[b * H + k] : g[b * 256 + (k - H)];
        s += x * gw[k * 256 + j];
    }
    gout[b * 256 + j] = g[b * 256 + j] + s;
}

// z = concat(h2[64], g[bid][256]) -> [N, 320]
__global__ void z_build(const float* __restrict__ h2, const float* __restrict__ g,
                        const int* __restrict__ bid, float* __restrict__ z, long total)
{
    long gidx = (long)blockIdx.x * blockDim.x + threadIdx.x;
    if (gidx >= total) return;
    int i = (int)(gidx / 320), c = (int)(gidx % 320);
    z[gidx] = (c < 64) ? h2[(long)i * 64 + c] : g[bid[i] * 256 + (c - 64)];
}

// final: a1[N,256] @ o2_w[256,11] + o2_b -> scatter into packed [16,4096,11]
__global__ void head_out(const float* __restrict__ a1, const float* __restrict__ w,
                         const float* __restrict__ b2, const int* __restrict__ bid,
                         const int* __restrict__ batches, float* __restrict__ out, int n)
{
    int gidx = blockIdx.x * blockDim.x + threadIdx.x;
    if (gidx >= n * 11) return;
    int i = gidx / 11, a = gidx % 11;
    float s = b2[a];
    const float* row = a1 + (long)i * 256;
    for (int k = 0; k < 256; ++k) s += row[k] * w[k * 11 + a];
    int bb = bid[i];
    int pos = i - batches[bb];
    out[((long)bb * 4096 + pos) * 11 + a] = s;
}

// ---------------------------------------------------------------------------
extern "C" void kernel_launch(void* const* d_in, const int* in_sizes, int n_in,
                              void* d_out, int out_size, void* d_ws, size_t ws_size,
                              hipStream_t stream) {
    const float* x        = (const float*)d_in[0];
    const int*   ei       = (const int*)d_in[1];
    const int*   batches  = (const int*)d_in[2];
    const float* conv1_w  = (const float*)d_in[4];
    const float* conv1_b  = (const float*)d_in[5];
    const float* conv2_w  = (const float*)d_in[6];
    const float* conv2_b  = (const float*)d_in[7];
    const float* g0_aw = (const float*)d_in[8];  const float* g0_ab = (const float*)d_in[9];
    const float* g0_fw = (const float*)d_in[10]; const float* g0_fb = (const float*)d_in[11];
    const float* g0_gw = (const float*)d_in[12]; const float* g0_gb = (const float*)d_in[13];
    const float* g1_aw = (const float*)d_in[14]; const float* g1_ab = (const float*)d_in[15];
    const float* g1_fw = (const float*)d_in[16]; const float* g1_fb = (const float*)d_in[17];
    const float* g1_gw = (const float*)d_in[18]; const float* g1_gb = (const float*)d_in[19];
    const float* g2_aw = (const float*)d_in[20]; const float* g2_ab = (const float*)d_in[21];
    const float* g2_fw = (const float*)d_in[22]; const float* g2_fb = (const float*)d_in[23];
    const float* g2_gw = (const float*)d_in[24]; const float* g2_gb = (const float*)d_in[25];
    const float* o1_w  = (const float*)d_in[26]; const float* o1_b  = (const float*)d_in[27];
    const float* o2_w  = (const float*)d_in[28]; const float* o2_b  = (const float*)d_in[29];
    float* out = (float*)d_out;

    // workspace carve-up (~309 MB)
    char* p = (char*)d_ws;
    int*   bid  = (int*)p;                 p += (size_t)NN * 4;
    float* dis  = (float*)p;               p += (size_t)NN * 4;
    float* att  = (float*)p;               p += (size_t)NN * 256 * 4;   // also h2
    float* feat = (float*)p;               p += (size_t)NN * 256 * 4;   // also att2
    float* hw   = (float*)p;               p += (size_t)NN * 256 * 4;   // also a1
    float* h    = (float*)p;               p += (size_t)NN * 256 * 4;   // h1, also feat2
    float* z    = (float*)p;               p += (size_t)NN * 320 * 4;
    float* gA   = (float*)p;               p += 16 * 256 * 4;
    float* gB   = (float*)p;               p += 16 * 256 * 4;
    float* ob   = (float*)p;               p += 16 * 256 * 4;

    const int T = 256;
    auto blocks = [](long c) { return (int)((c + 255) / 256); };

    // metadata
    compute_bid<<<blocks(NN), T, 0, stream>>>(batches, bid, NN);
    fill_kernel<<<blocks(NN), T, 0, stream>>>(dis, 1.0f, NN);            // self loop
    deg_count<<<blocks(EE), T, 0, stream>>>(ei, dis, EE);
    deg_finalize<<<blocks(NN), T, 0, stream>>>(dis, NN);
    fill_kernel<<<blocks(16 * 256), T, 0, stream>>>(gA, 0.0f, 16 * 256); // g = 0

    dim3 g256(NN / 128, 256 / 64), g64(NN / 128, 1);

    // ---- attn0 (v = x, H=256) ----
    wmma_gemm<<<g256, T, 0, stream>>>(x, g0_aw, g0_ab, att, NN, 256, 128, 2);
    wmma_gemm<<<g256, T, 0, stream>>>(x, g0_fw, g0_fb, feat, NN, 256, 128, 2);
    fill_kernel<<<blocks(16 * 256), T, 0, stream>>>(ob, 0.0f, 16 * 256);
    attn_reduce<256><<<NN / 32, T, 0, stream>>>(att, feat, bid, ob, NN);
    g_update<<<16, 256, 0, stream>>>(ob, gA, g0_gw, g0_gb, gB, 256);     // g -> gB

    // ---- gcn1: h = relu(Ahat (x@W1) + b1) ----
    wmma_gemm<<<g256, T, 0, stream>>>(x, conv1_w, nullptr, hw, NN, 256, 128, 0);
    gcn_self<<<blocks((long)NN * 256), T, 0, stream>>>(hw, dis, h, 256, (long)NN * 256);
    gcn_edge<256><<<EE, T, 0, stream>>>(ei, hw, dis, h, EE);
    gcn_fin<<<blocks((long)NN * 256), T, 0, stream>>>(h, conv1_b, 256, (long)NN * 256);

    // ---- attn1 (v = h, H=256) ----
    wmma_gemm<<<g256, T, 0, stream>>>(h, g1_aw, g1_ab, att, NN, 256, 256, 2);
    wmma_gemm<<<g256, T, 0, stream>>>(h, g1_fw, g1_fb, feat, NN, 256, 256, 2);
    fill_kernel<<<blocks(16 * 256), T, 0, stream>>>(ob, 0.0f, 16 * 256);
    attn_reduce<256><<<NN / 32, T, 0, stream>>>(att, feat, bid, ob, NN);
    g_update<<<16, 256, 0, stream>>>(ob, gB, g1_gw, g1_gb, gA, 256);     // g -> gA

    // ---- gcn2: h2 = relu(Ahat (h@W2) + b2), stored in `att` buffer ----
    wmma_gemm<<<g64, T, 0, stream>>>(h, conv2_w, nullptr, hw, NN, 64, 256, 0);
    gcn_self<<<blocks((long)NN * 64), T, 0, stream>>>(hw, dis, att, 64, (long)NN * 64);
    gcn_edge<64><<<(EE + 3) / 4, T, 0, stream>>>(ei, hw, dis, att, EE);
    gcn_fin<<<blocks((long)NN * 64), T, 0, stream>>>(att, conv2_b, 64, (long)NN * 64);
    float* h2 = att;

    // ---- attn2 (v = h2, H=64): att2 in `feat`, feat2 in `h` ----
    wmma_gemm<<<g64, T, 0, stream>>>(h2, g2_aw, g2_ab, feat, NN, 64, 64, 2);
    wmma_gemm<<<g64, T, 0, stream>>>(h2, g2_fw, g2_fb, h, NN, 64, 64, 2);
    fill_kernel<<<blocks(16 * 64), T, 0, stream>>>(ob, 0.0f, 16 * 64);
    attn_reduce<64><<<NN / 32, T, 0, stream>>>(feat, h, bid, ob, NN);
    g_update<<<16, 256, 0, stream>>>(ob, gA, g2_gw, g2_gb, gB, 64);      // g -> gB

    // ---- head ----
    z_build<<<blocks((long)NN * 320), T, 0, stream>>>(h2, gB, bid, z, (long)NN * 320);
    wmma_gemm<<<g256, T, 0, stream>>>(z, o1_w, o1_b, hw, NN, 256, 320, 3); // relu+bias
    fill_kernel<<<blocks((long)16 * 4096 * 11), T, 0, stream>>>(out, INF_NEG, (long)16 * 4096 * 11);
    head_out<<<blocks((long)NN * 11), T, 0, stream>>>(hw, o2_w, o2_b, bid, batches, out, NN);
}
